// NeuronMiniMaxM2DecoderLayer_2671469658749
// MI455X (gfx1250) — matrix operand order repaired and checked by reading, not compiled, not run on hardware
//
#include <hip/hip_runtime.h>
#include <math.h>

typedef __bf16 bf16;
typedef __attribute__((ext_vector_type(16))) __bf16 v16bf;
typedef __attribute__((ext_vector_type(8)))  __bf16 v8bf;
typedef __attribute__((ext_vector_type(8)))  float  v8f;
typedef __attribute__((ext_vector_type(4)))  unsigned int u32x4;
typedef __attribute__((ext_vector_type(8)))  int i32x8;
typedef __attribute__((ext_vector_type(4)))  int i32x4;

#if defined(__has_builtin)
#if __has_builtin(__builtin_amdgcn_tensor_load_to_lds) && (__clang_major__ < 23)
#define USE_TDM 1
#endif
#endif
#ifndef USE_TDM
#define USE_TDM 0
#endif

static __device__ __forceinline__ v8f wmma_bf16(v16bf a, v16bf b, v8f c) {
  // D(f32 16x16) = A(bf16 16x32) * B(bf16 32x16) + C
  return __builtin_amdgcn_wmma_f32_16x16x32_bf16(false, a, false, b,
                                                 (short)0, c, false, false);
}

static __device__ __forceinline__ v16bf ldg32(const bf16* p) {
  v8bf lo = *(const v8bf*)p;
  v8bf hi = *(const v8bf*)(p + 8);
  return __builtin_shufflevector(lo, hi,
      0, 1, 2, 3, 4, 5, 6, 7, 8, 9, 10, 11, 12, 13, 14, 15);
}
static __device__ __forceinline__ v16bf ldg_split(const bf16* p) {   // 8 at p, 8 at p+16
  v8bf lo = *(const v8bf*)p;
  v8bf hi = *(const v8bf*)(p + 16);
  return __builtin_shufflevector(lo, hi,
      0, 1, 2, 3, 4, 5, 6, 7, 8, 9, 10, 11, 12, 13, 14, 15);
}

#if USE_TDM
// Issue a 2-D TDM tile load: tile_w contiguous elems x tile_h rows of 2-byte
// data, row stride (elems) within the source tensor, into LDS at lds_off.
static __device__ __forceinline__ void tdm_load_2d(unsigned lds_off,
                                                   const void* gtile,
                                                   unsigned tile_w, unsigned tile_h,
                                                   unsigned long long row_stride,
                                                   unsigned tensor_w, unsigned tensor_h) {
  unsigned long long ga = (unsigned long long)(size_t)gtile;
  u32x4 g0;
  g0[0] = 1u;                                        // count=1, user descriptor
  g0[1] = lds_off;                                   // lds_addr (bytes)
  g0[2] = (unsigned)(ga & 0xffffffffu);              // global_addr[31:0]
  g0[3] = (unsigned)((ga >> 32) & 0x01ffffffu) | (2u << 30);  // addr[56:32]|type=2
  i32x8 g1;
  g1[0] = (int)(1u << 16);                           // wg_mask=0, data_size=1 (2B)
  g1[1] = (int)((tensor_w & 0xffffu) << 16);         // tensor_dim0[15:0]
  g1[2] = (int)((tensor_w >> 16) | ((tensor_h & 0xffffu) << 16));
  g1[3] = (int)((tensor_h >> 16) | ((tile_w & 0xffffu) << 16));
  g1[4] = (int)(tile_h & 0xffffu);                   // tile_dim1 (tile_dim2=0)
  g1[5] = (int)(row_stride & 0xffffffffu);           // tensor_dim0_stride[31:0]
  g1[6] = (int)((row_stride >> 32) & 0xffffu);       // stride[47:32], dim1_stride=0
  g1[7] = 0;
  i32x4 z; z[0] = 0; z[1] = 0; z[2] = 0; z[3] = 0;   // groups 2/3 unused (2-D)
  __builtin_amdgcn_tensor_load_to_lds(g0, g1, z, z, 0);
}
#endif

// ---------------- elementwise helpers ----------------

__global__ void k_cvt_bf16(const float* __restrict__ in, bf16* __restrict__ out, size_t n) {
  size_t i = (size_t)blockIdx.x * 256 + threadIdx.x;
  if (i < n) out[i] = (bf16)in[i];
}

__global__ void k_copy_f32(const float* __restrict__ a, float* __restrict__ out, size_t n) {
  size_t i = (size_t)blockIdx.x * 256 + threadIdx.x;
  if (i < n) out[i] = a[i];
}

__global__ void k_silu_mul(const float* __restrict__ g, const float* __restrict__ u,
                           bf16* __restrict__ h, size_t n) {
  size_t i = (size_t)blockIdx.x * 256 + threadIdx.x;
  if (i < n) {
    float gv = g[i];
    h[i] = (bf16)((gv / (1.f + __expf(-gv))) * u[i]);
  }
}

// v [B,S,NH,D] f32  ->  vt [B,NH,D,S] bf16
__global__ void k_transpose_v(const float* __restrict__ v, bf16* __restrict__ vt,
                              int S, int NH, int D, size_t n) {
  size_t i = (size_t)blockIdx.x * 256 + threadIdx.x;
  if (i >= n) return;
  size_t s = i % S;  size_t r = i / S;
  size_t d = r % D;  r /= D;
  size_t h = r % NH; size_t b = r / NH;
  vt[i] = (bf16)v[((b * S + s) * NH + h) * D + d];
}

// ---------------- RMSNorm (row = token, H elems) -> bf16 ----------------

__global__ __launch_bounds__(256)
void k_rmsnorm(const float* __restrict__ x, const float* __restrict__ w,
               bf16* __restrict__ out, int H) {
  __shared__ float red[256];
  int row = blockIdx.x, tid = threadIdx.x;
  size_t base = (size_t)row * H;
  float s = 0.f;
  for (int i = tid; i < H; i += 256) { float v = x[base + i]; s += v * v; }
  red[tid] = s; __syncthreads();
  for (int k = 128; k > 0; k >>= 1) { if (tid < k) red[tid] += red[tid + k]; __syncthreads(); }
  float inv = rsqrtf(red[0] / (float)H + 1e-6f);
  for (int i = tid; i < H; i += 256)
    out[base + i] = (bf16)(x[base + i] * inv * w[i]);
}

// ---------------- per-head RMSNorm + RoPE -> bf16 (scale folds 1/sqrt(D) for q)

__global__ __launch_bounds__(128)
void k_qknorm_rope(const float* __restrict__ x, const float* __restrict__ w,
                   bf16* __restrict__ out, int S, int NH, int D, float scale) {
  __shared__ float red[128];
  int row = blockIdx.x;          // over B*S*NH
  int tid = threadIdx.x;         // D threads
  size_t base = (size_t)row * D;
  float v = x[base + tid];
  red[tid] = v * v;
  __syncthreads();
  for (int k = D >> 1; k > 0; k >>= 1) { if (tid < k) red[tid] += red[tid + k]; __syncthreads(); }
  float inv = rsqrtf(red[0] / (float)D + 1e-6f);
  int half = D >> 1;
  if (tid < half) {
    int s_pos = (row / NH) % S;
    float x1 = x[base + tid]        * inv * w[tid];
    float x2 = x[base + tid + half] * inv * w[tid + half];
    float ang = (float)s_pos * __powf(10000.f, -(float)tid / (float)half);
    float c = __cosf(ang), sn = __sinf(ang);
    out[base + tid]        = (bf16)((x1 * c - x2 * sn) * scale);
    out[base + tid + half] = (bf16)((x2 * c + x1 * sn) * scale);
  }
}

// ---------------- WMMA bf16 GEMM with fused epilogue ----------------
// MODE 0: C = A@B      MODE 1: C = R + A@B      MODE 2: C += rowScale * A@B
// Block 256 thr (8 waves), tile 128x64; wave grid 4x2, 32x32 per wave
// (2x2 WMMA accumulators => 4 v_wmma per wave per 32-wide K step).
// A tile staged into LDS by the Tensor Data Mover when available.

template <int MODE>
__global__ __launch_bounds__(256)
void k_gemm_bf16(const bf16* __restrict__ A, const bf16* __restrict__ Bm,
                 float* __restrict__ C, const float* __restrict__ R,
                 int M, int N, int K,
                 const float* __restrict__ rowScale, int rsStride) {
  __shared__ alignas(32) bf16 lA[128 * 32];   // [m][k], pitch 32 (TDM layout)
  __shared__ alignas(32) bf16 lBt[64 * 40];   // [n][k] transposed, pitch 40
  const int tid  = threadIdx.x;
  const int wave = tid >> 5, lane = tid & 31;
  const int hi = lane >> 4, lo = lane & 15;
  const int m0 = blockIdx.y * 128, n0 = blockIdx.x * 64;
  const int mw = (wave & 3) * 32, nw = (wave >> 2) * 32;

  v8f acc[2][2];
  for (int i = 0; i < 2; ++i)
    for (int j = 0; j < 2; ++j)
      for (int r = 0; r < 8; ++r) acc[i][j][r] = 0.f;

  const int arow0 = mw + lo;
  const int akb   = hi * 16;
  const int bcol0 = nw + lo;
  const int bkb   = hi * 16;
#if USE_TDM
  const unsigned ldsA_off = (unsigned)(size_t)(void*)lA;
  const bf16* gA = A + (size_t)m0 * K;          // bumped by 32 per step
#else
  const int sr = tid >> 2, sc = (tid & 3) * 8;  // A staging coords
  const bf16* gA0 = A + (size_t)(m0 + sr) * K + sc;
  const bf16* gA1 = A + (size_t)(m0 + sr + 64) * K + sc;
#endif
  const int skk = tid >> 3, sn8 = (tid & 7) * 8;   // B staging coords
  const bf16* gB = Bm + (size_t)skk * N + n0 + sn8;

  for (int k0 = 0; k0 < K; k0 += 32) {
#if USE_TDM
    if (wave == 0) {   // single DMA for the whole 128x32 A tile
      tdm_load_2d(ldsA_off, gA, 32u, 128u,
                  (unsigned long long)K, (unsigned)K, (unsigned)M);
      __builtin_amdgcn_s_wait_tensorcnt(0);
    }
    gA += 32;
#else
    *(v8bf*)&lA[sr * 32 + sc]        = *(const v8bf*)gA0;
    *(v8bf*)&lA[(sr + 64) * 32 + sc] = *(const v8bf*)gA1;
    if (k0 + 32 < K) {
      __builtin_prefetch(gA0 + 32, 0, 1);
      __builtin_prefetch(gA1 + 32, 0, 1);
    }
    gA0 += 32; gA1 += 32;
#endif
    {
      // B tile 32(K) x 64(N): one 16B chunk per thread, stored transposed
      v8bf row = *(const v8bf*)gB;
      #pragma unroll
      for (int j = 0; j < 8; ++j) lBt[(sn8 + j) * 40 + skk] = row[j];
      if (k0 + 32 < K) __builtin_prefetch(gB + (size_t)32 * N, 0, 1);
      gB += (size_t)32 * N;
    }
    __syncthreads();

    v16bf bfrag[2];
    #pragma unroll
    for (int ni = 0; ni < 2; ++ni)
      bfrag[ni] = ldg32(&lBt[(size_t)(bcol0 + ni * 16) * 40 + bkb]);
    #pragma unroll
    for (int mi = 0; mi < 2; ++mi) {
      v16bf afrag = ldg32(&lA[(size_t)(arow0 + mi * 16) * 32 + akb]);
      #pragma unroll
      for (int ni = 0; ni < 2; ++ni)
        acc[mi][ni] = wmma_bf16(afrag, bfrag[ni], acc[mi][ni]);
    }
    __syncthreads();
  }

  #pragma unroll
  for (int mi = 0; mi < 2; ++mi) {
    int rbase = m0 + mw + mi * 16 + hi * 8;
    #pragma unroll
    for (int r = 0; r < 8; ++r) {
      int grow = rbase + r;
      float rs = 1.f;
      if (MODE == 2) rs = rowScale[(size_t)grow * rsStride];
      #pragma unroll
      for (int ni = 0; ni < 2; ++ni) {
        int gcol = n0 + nw + ni * 16 + lo;
        size_t ci = (size_t)grow * N + gcol;
        if (MODE == 0)      C[ci]  = acc[mi][ni][r];
        else if (MODE == 1) C[ci]  = R[ci] + acc[mi][ni][r];
        else                C[ci] += acc[mi][ni][r] * rs;
      }
    }
  }
}

// ---------------- flash attention (D==128), per-wave 16-query tile ----------------
// Scores computed transposed: S^T(32keys x 16q) = K_tile(32xD) @ Q^T(Dx16),
// so exp(P) accumulators are already in WMMA A-fragment layout for P@V.
// Output written directly as bf16 (feeds the Wo WMMA GEMM).

__global__ __launch_bounds__(128)
void k_flash_attn(const bf16* __restrict__ Q, const bf16* __restrict__ Kb,
                  const bf16* __restrict__ Vt, bf16* __restrict__ O,
                  int S, int NH, int D) {
  int wave = threadIdx.x >> 5, lane = threadIdx.x & 31;
  int b = blockIdx.z, h = blockIdx.y;
  int qbase = blockIdx.x * 64 + wave * 16;
  int hi = lane >> 4, lo = lane & 15;
  int q = qbase + lo;

  v16bf qf[4];                                 // Q^T B-fragments, K-dim = D
  {
    const bf16* p = &Q[((size_t)(b * S + q) * NH + h) * D + hi * 8];
    #pragma unroll
    for (int kk = 0; kk < 4; ++kk) qf[kk] = ldg_split(p + kk * 32);
  }

  v8f Oacc[8];
  for (int c = 0; c < 8; ++c) for (int r = 0; r < 8; ++r) Oacc[c][r] = 0.f;
  float m_run = -3.0e38f, l_run = 0.f;
  const int kend = qbase + 16;                 // causal: keys <= qbase+15

  // strength-reduced pointers (bumped by constants per key-tile)
  const size_t rstride = (size_t)NH * D;                  // elems per key row
  const bf16* kp0 = Kb + ((size_t)(b * S + lo) * NH + h) * D + hi * 8;
  const bf16* kp1 = kp0 + 16 * rstride;
  const bf16* vp0 = Vt + ((size_t)(b * NH + h) * D + lo) * S + hi * 8;

  for (int kb = 0; kb < kend; kb += 32) {
    v8f s0, s1;
    for (int r = 0; r < 8; ++r) { s0[r] = 0.f; s1[r] = 0.f; }
    #pragma unroll
    for (int kk = 0; kk < 4; ++kk) {
      v16bf a0 = ldg_split(kp0 + kk * 32);
      v16bf a1 = ldg_split(kp1 + kk * 32);
      s0 = wmma_bf16(a0, qf[kk], s0);
      s1 = wmma_bf16(a1, qf[kk], s1);
    }
    // causal mask (acc element: key = kb + r + 8*hi (+16 for s1), query = q)
    #pragma unroll
    for (int r = 0; r < 8; ++r) {
      int key0 = kb + r + hi * 8;
      if (key0 > q)      s0[r] = -1e30f;
      if (key0 + 16 > q) s1[r] = -1e30f;
    }
    // per-query (lane-pair) online softmax
    float mt = -3.0e38f;
    for (int r = 0; r < 8; ++r) mt = fmaxf(mt, fmaxf(s0[r], s1[r]));
    mt = fmaxf(mt, __shfl_xor(mt, 16));
    float mnew  = fmaxf(m_run, mt);
    float alpha = __expf(m_run - mnew);
    float ls = 0.f;
    #pragma unroll
    for (int r = 0; r < 8; ++r) {
      s0[r] = __expf(s0[r] - mnew); ls += s0[r];
      s1[r] = __expf(s1[r] - mnew); ls += s1[r];
    }
    ls += __shfl_xor(ls, 16);
    l_run = l_run * alpha + ls;
    m_run = mnew;
    // rescale O (O rows = queries -> register-indexed alpha via bpermute)
    #pragma unroll
    for (int r = 0; r < 8; ++r) {
      float ar = __shfl(alpha, r + hi * 8);
      #pragma unroll
      for (int c = 0; c < 8; ++c) Oacc[c][r] *= ar;
    }
    // exp(P) is already A-fragment-shaped: pack to bf16, do P(16x32) @ V(32x16c)
    v16bf pa;
    #pragma unroll
    for (int j = 0; j < 8; ++j) { pa[j] = (bf16)s0[j]; pa[8 + j] = (bf16)s1[j]; }
    #pragma unroll
    for (int c = 0; c < 8; ++c) {
      v16bf vb = ldg_split(vp0 + (size_t)c * 16 * S + kb);
      Oacc[c] = wmma_bf16(pa, vb, Oacc[c]);
    }
    kp0 += 32 * rstride;
    kp1 += 32 * rstride;
  }
  #pragma unroll
  for (int r = 0; r < 8; ++r) {
    float lr  = __shfl(l_run, r + hi * 8);
    float inv = 1.f / lr;
    int qrow  = qbase + r + hi * 8;
    #pragma unroll
    for (int c = 0; c < 8; ++c)
      O[((size_t)(b * S + qrow) * NH + h) * D + c * 16 + lo] = (bf16)(Oacc[c][r] * inv);
  }
}

// ---------------- router: wave per token, sigmoid + bias-corrected top-2 ----------------

__global__ __launch_bounds__(256)
void k_router(const bf16* __restrict__ xt, const float* __restrict__ Wr,
              const float* __restrict__ bias, float* __restrict__ comb,
              int T, int H, int E) {
  int wave = threadIdx.x >> 5, lane = threadIdx.x & 31;
  int t = blockIdx.x * 8 + wave;
  if (t >= T) return;
  float acc[8];
  for (int e = 0; e < 8; ++e) acc[e] = 0.f;
  for (int hh = lane; hh < H; hh += 32) {
    float x = (float)xt[(size_t)t * H + hh];
    for (int e = 0; e < E; ++e) acc[e] += x * Wr[(size_t)hh * E + e];
  }
  for (int e = 0; e < E; ++e)
    for (int m = 16; m >= 1; m >>= 1) acc[e] += __shfl_xor(acc[e], m);
  if (lane == 0) {
    float sig[8], sel[8];
    for (int e = 0; e < E; ++e) { sig[e] = 1.f / (1.f + __expf(-acc[e])); sel[e] = sig[e] + bias[e]; }
    int i0 = 0;
    for (int e = 1; e < E; ++e) if (sel[e] > sel[i0]) i0 = e;
    int i1 = -1;
    for (int e = 0; e < E; ++e) { if (e == i0) continue; if (i1 < 0 || sel[e] > sel[i1]) i1 = e; }
    float w0 = sig[i0], w1 = sig[i1], sum = w0 + w1;
    for (int e = 0; e < E; ++e) comb[(size_t)t * E + e] = 0.f;
    comb[(size_t)t * E + i0] = w0 / sum;
    comb[(size_t)t * E + i1] = w1 / sum;
  }
}

// ---------------- host orchestration ----------------

#define CEILDIV(a, b) (((a) + (b) - 1) / (b))

extern "C" void kernel_launch(void* const* d_in, const int* in_sizes, int n_in,
                              void* d_out, int out_size, void* d_ws, size_t ws_size,
                              hipStream_t stream) {
  (void)n_in; (void)out_size; (void)ws_size;
  const float* hs    = (const float*)d_in[0];
  const float* ln1_w = (const float*)d_in[1];
  const float* ln2_w = (const float*)d_in[2];
  const float* qn_w  = (const float*)d_in[3];
  const float* kn_w  = (const float*)d_in[4];
  const float* Wq    = (const float*)d_in[5];
  const float* Wk    = (const float*)d_in[6];
  const float* Wv    = (const float*)d_in[7];
  const float* Wo    = (const float*)d_in[8];
  const float* Wr    = (const float*)d_in[9];
  const float* rbias = (const float*)d_in[10];
  const float* Wg    = (const float*)d_in[11];
  const float* Wu    = (const float*)d_in[12];
  const float* Wd    = (const float*)d_in[13];

  const int H   = in_sizes[1];
  const int D   = in_sizes[3];
  const int NHD = in_sizes[5] / H;
  const int NH  = NHD / D;
  const int E   = in_sizes[10];
  const int I   = in_sizes[11] / (E * H);
  const int T   = in_sizes[0] / H;
  const int Bb  = 2;                 // batch (fixed by reference setup)
  const int S   = T / Bb;

  char* ws = (char*)d_ws; size_t off = 0;
  auto alloc = [&](size_t bytes) -> void* {
    void* p = ws + off; off += (bytes + 255) & ~(size_t)255; return p;
  };

  const size_t nW  = (size_t)H * NHD;      // proj weight elems
  const size_t nE  = (size_t)E * H * I;    // expert weight elems
  const size_t nTH = (size_t)T * H;
  const size_t nTN = (size_t)T * NHD;
  const size_t nTI = (size_t)T * I;

  bf16* Wq_b = (bf16*)alloc(nW * 2);
  bf16* Wk_b = (bf16*)alloc(nW * 2);
  bf16* Wv_b = (bf16*)alloc(nW * 2);
  bf16* Wo_b = (bf16*)alloc(nW * 2);
  bf16* Wg_b = (bf16*)alloc(nE * 2);
  bf16* Wu_b = (bf16*)alloc(nE * 2);
  bf16* Wd_b = (bf16*)alloc(nE * 2);
  bf16* xnrm = (bf16*)alloc(nTH * 2);
  float* qf  = (float*)alloc(nTN * 4);
  float* kf  = (float*)alloc(nTN * 4);
  float* vf  = (float*)alloc(nTN * 4);
  bf16* qhat = (bf16*)alloc(nTN * 2);
  bf16* khat = (bf16*)alloc(nTN * 2);
  bf16* vt   = (bf16*)alloc(nTN * 2);
  bf16* attnb = (bf16*)alloc(nTN * 2);
  float* hidden = (float*)alloc(nTH * 4);
  bf16* xt_b  = (bf16*)alloc(nTH * 2);
  float* comb = (float*)alloc((size_t)T * E * 4);
  bf16* hexp  = (bf16*)alloc(nTI * 2);
  // reuse big fp32 buffers after their producers are consumed
  float* gbuf  = vf;                       // expert gate (vf free after transpose)
  float* ubuf  = vf + nTI;                 // expert up (T*I + T*I == T*NHD floats)

  // weight conversion fp32 -> bf16
  k_cvt_bf16<<<CEILDIV(nW, 256), 256, 0, stream>>>(Wq, Wq_b, nW);
  k_cvt_bf16<<<CEILDIV(nW, 256), 256, 0, stream>>>(Wk, Wk_b, nW);
  k_cvt_bf16<<<CEILDIV(nW, 256), 256, 0, stream>>>(Wv, Wv_b, nW);
  k_cvt_bf16<<<CEILDIV(nW, 256), 256, 0, stream>>>(Wo, Wo_b, nW);
  k_cvt_bf16<<<CEILDIV(nE, 256), 256, 0, stream>>>(Wg, Wg_b, nE);
  k_cvt_bf16<<<CEILDIV(nE, 256), 256, 0, stream>>>(Wu, Wu_b, nE);
  k_cvt_bf16<<<CEILDIV(nE, 256), 256, 0, stream>>>(Wd, Wd_b, nE);

  // ---- attention block ----
  k_rmsnorm<<<T, 256, 0, stream>>>(hs, ln1_w, xnrm, H);
  dim3 gProj(NHD / 64, T / 128);
  k_gemm_bf16<0><<<gProj, 256, 0, stream>>>(xnrm, Wq_b, qf, nullptr, T, NHD, H, nullptr, 0);
  k_gemm_bf16<0><<<gProj, 256, 0, stream>>>(xnrm, Wk_b, kf, nullptr, T, NHD, H, nullptr, 0);
  k_gemm_bf16<0><<<gProj, 256, 0, stream>>>(xnrm, Wv_b, vf, nullptr, T, NHD, H, nullptr, 0);

  float qscale = 1.0f / sqrtf((float)D);
  k_qknorm_rope<<<(size_t)T * NH, D, 0, stream>>>(qf, qn_w, qhat, S, NH, D, qscale);
  k_qknorm_rope<<<(size_t)T * NH, D, 0, stream>>>(kf, kn_w, khat, S, NH, D, 1.0f);
  k_transpose_v<<<CEILDIV(nTN, 256), 256, 0, stream>>>(vf, vt, S, NH, D, nTN);

  k_flash_attn<<<dim3(S / 64, NH, Bb), 128, 0, stream>>>(qhat, khat, vt, attnb, S, NH, D);

  // hidden = hs + attn @ Wo   (residual fused into GEMM epilogue)
  k_gemm_bf16<1><<<dim3(H / 64, T / 128), 256, 0, stream>>>(attnb, Wo_b, hidden, hs,
                                                            T, H, NHD, nullptr, 0);

  // ---- MoE block ----
  k_rmsnorm<<<T, 256, 0, stream>>>(hidden, ln2_w, xt_b, H);
  k_router<<<CEILDIV(T, 8), 256, 0, stream>>>(xt_b, Wr, rbias, comb, T, H, E);
  k_copy_f32<<<CEILDIV(nTH, 256), 256, 0, stream>>>(hidden, (float*)d_out, nTH);

  dim3 gUp(I / 64, T / 128), gDown(H / 64, T / 128);
  for (int e = 0; e < E; ++e) {
    const bf16* Wg_e = Wg_b + (size_t)e * H * I;
    const bf16* Wu_e = Wu_b + (size_t)e * H * I;
    const bf16* Wd_e = Wd_b + (size_t)e * I * H;
    k_gemm_bf16<0><<<gUp, 256, 0, stream>>>(xt_b, Wg_e, gbuf, nullptr, T, I, H, nullptr, 0);
    k_gemm_bf16<0><<<gUp, 256, 0, stream>>>(xt_b, Wu_e, ubuf, nullptr, T, I, H, nullptr, 0);
    k_silu_mul<<<CEILDIV(nTI, 256), 256, 0, stream>>>(gbuf, ubuf, hexp, nTI);
    // d_out[t,:] += comb[t,e] * (hexp @ Wd_e)
    k_gemm_bf16<2><<<gDown, 256, 0, stream>>>(hexp, Wd_e, (float*)d_out, nullptr,
                                              T, H, I, comb + e, E);
  }
}
